// VectorizedInteraction_33947421508010
// MI455X (gfx1250) — matrix operand order repaired and checked by reading, not compiled, not run on hardware
//
#include <hip/hip_runtime.h>
#include <hip/hip_bf16.h>

// Problem constants (match reference)
#define BB 2
#define TT 512
#define NF 6
#define DD 32
#define HIDN 128
#define NHEAD 4
#define HDIM 32
#define GHID 32
#define FDIM 128
#define LL (TT * NF)          // 3072
#define EPSN 1e-5f
// 1/sqrt(32) * log2(e): scores land directly in exp2 domain (softmax is
// base-invariant), so the inner loop uses raw v_exp_f32 with no mul.
#define QSCALE_LOG2E 0.25503486373f
#define NSPLIT 4              // waves per query tile (split-K over keys)

typedef __attribute__((ext_vector_type(16))) _Float16 v16h;
typedef __attribute__((ext_vector_type(8)))  float    v8f;

struct alignas(16) h8 { _Float16 h[8]; };
union H16 { v16h v; h8 part[2]; _Float16 h[16]; unsigned u[8]; };

// raw v_exp_f32 (exp2), no OCML denormal guard -- fine for softmax weights
__device__ __forceinline__ float fast_exp2(float x) {
  return __builtin_amdgcn_exp2f(x);
}

__device__ __forceinline__ float wsum(float v) {
  #pragma unroll
  for (int m = 16; m >= 1; m >>= 1) v += __shfl_xor(v, m, 32);
  return v;
}

// ---------------------------------------------------------------------------
// Kernel 1: gater + gated QKV projections.  One wave (32 lanes) per token row.
// QKV + gater weights staged in LDS once per 8-wave block.
// Emits f16 Q (pre-scaled by gate/sqrt(HD)*log2e), f16 K (gated),
// f16 V^T (gated, [b,h,d,l] layout -> contiguous A-tile loads in attention).
// ---------------------------------------------------------------------------
__global__ __launch_bounds__(256) void prep_kernel(
    const float* __restrict__ x,
    const float* __restrict__ glnw, const float* __restrict__ glnb,
    const float* __restrict__ gw1,  const float* __restrict__ gb1,
    const float* __restrict__ gw2,  const float* __restrict__ gb2,
    const float* __restrict__ qw,   const float* __restrict__ qb,
    const float* __restrict__ kw,   const float* __restrict__ kbias,
    const float* __restrict__ vw,   const float* __restrict__ vb,
    _Float16* __restrict__ Qh, _Float16* __restrict__ Kh,
    _Float16* __restrict__ VT)
{
  __shared__ float s_qw[DD * HIDN];
  __shared__ float s_kw[DD * HIDN];
  __shared__ float s_vw[DD * HIDN];
  __shared__ float s_gw1[DD * GHID];

  for (int i = threadIdx.x; i < DD * HIDN; i += 256) {
    s_qw[i] = qw[i];
    s_kw[i] = kw[i];
    s_vw[i] = vw[i];
  }
  for (int i = threadIdx.x; i < DD * GHID; i += 256) s_gw1[i] = gw1[i];
  __syncthreads();

  const int lane = threadIdx.x & 31;
  const int row  = blockIdx.x * 8 + (threadIdx.x >> 5);   // [0, B*L)
  const int b = row / LL;
  const int l = row % LL;

  // LayerNorm over D=32 (one element per lane)
  const float xv = x[(size_t)row * DD + lane];
  const float mu = wsum(xv) * (1.f / DD);
  const float d0 = xv - mu;
  const float var = wsum(d0 * d0) * (1.f / DD);
  const float xn = d0 * rsqrtf(var + EPSN) * glnw[lane] + glnb[lane];

  // gater hidden + QKV accumulators via shfl-broadcast dot products
  float hacc = gb1[lane];
  float aq[4], ak[4], av[4];
  #pragma unroll
  for (int i = 0; i < 4; ++i) {
    aq[i] = qb[lane + 32 * i];
    ak[i] = kbias[lane + 32 * i];
    av[i] = vb[lane + 32 * i];
  }
  for (int k = 0; k < DD; ++k) {
    const float xk  = __shfl(xv, k, 32);   // raw x feeds QKV
    const float xnk = __shfl(xn, k, 32);   // normalized x feeds gater
    hacc += xnk * s_gw1[k * GHID + lane];
    #pragma unroll
    for (int i = 0; i < 4; ++i) {
      const int c = k * HIDN + lane + 32 * i;
      aq[i] += xk * s_qw[c];
      ak[i] += xk * s_kw[c];
      av[i] += xk * s_vw[c];
    }
  }
  const float hsil = hacc / (1.f + __expf(-hacc));           // SiLU
  const float gsum = wsum(hsil * gw2[lane]) + gb2[0];
  const float g = 1.f / (1.f + __expf(-gsum));               // sigmoid gate

  #pragma unroll
  for (int i = 0; i < 4; ++i) {   // i == head (col = lane + 32*i, d = lane)
    const size_t qk_idx = ((size_t)(b * NHEAD + i) * LL + l) * HDIM + lane;
    Qh[qk_idx] = (_Float16)(aq[i] * g * QSCALE_LOG2E);
    Kh[qk_idx] = (_Float16)(ak[i] * g);
    VT[((size_t)(b * NHEAD + i) * HDIM + lane) * LL + l] = (_Float16)(av[i] * g);
  }
}

// ---------------------------------------------------------------------------
// One 32-key flash-attention step (transposed formulation, exp2 domain).
// MASKED=false: fast path, every key visible to every lane's query.
// MASKED=true : boundary steps, per-lane horizon test kj < jlane
//               (hoisted form of kj/NF <= qt -- no integer division).
// ---------------------------------------------------------------------------
template <bool MASKED>
__device__ __forceinline__ void attn_step(
    int c, int qsub, int hi, int jlane,
    const _Float16* __restrict__ kbase, const _Float16* __restrict__ vtb,
    v16h bqv, float& m_run, float& l_run, v8f& o1, v8f& o2)
{
  const v8f zero = {};

  // A-matrix K tiles: lane m = key token, elems = dim subsets per ISA layout
  const _Float16* k1 = kbase + (size_t)(c + qsub) * HDIM + 8 * hi;
  const _Float16* k2 = k1 + 16 * HDIM;
  H16 ak1, ak2;
  ak1.part[0] = *(const h8*)k1;  ak1.part[1] = *(const h8*)(k1 + 16);
  ak2.part[0] = *(const h8*)k2;  ak2.part[1] = *(const h8*)(k2 + 16);

  __builtin_prefetch((const void*)(k1 + 32 * HDIM), 0, 1);

  // S^T = K * Q^T  (f32 accumulate); scores already in log2 domain
  v8f s1 = __builtin_amdgcn_wmma_f32_16x16x32_f16(false, ak1.v, false, bqv,
                                                  (short)0, zero, false, false);
  v8f s2 = __builtin_amdgcn_wmma_f32_16x16x32_f16(false, ak2.v, false, bqv,
                                                  (short)0, zero, false, false);

  float e1[8], e2[8];
  if (MASKED) {
    const int base = c + 8 * hi;            // key index of s?[0]
    #pragma unroll
    for (int gg = 0; gg < 8; ++gg) {
      e1[gg] = (base + gg      < jlane) ? s1[gg] : -1e30f;
      e2[gg] = (base + gg + 16 < jlane) ? s2[gg] : -1e30f;
    }
  } else {
    #pragma unroll
    for (int gg = 0; gg < 8; ++gg) { e1[gg] = s1[gg]; e2[gg] = s2[gg]; }
  }

  // tree max (short dependency chain; WMMA co-executes with this VALU)
  const float ma = fmaxf(fmaxf(e1[0], e1[1]), fmaxf(e1[2], e1[3]));
  const float mb = fmaxf(fmaxf(e1[4], e1[5]), fmaxf(e1[6], e1[7]));
  const float mc = fmaxf(fmaxf(e2[0], e2[1]), fmaxf(e2[2], e2[3]));
  const float md = fmaxf(fmaxf(e2[4], e2[5]), fmaxf(e2[6], e2[7]));
  float mloc = fmaxf(fmaxf(ma, mb), fmaxf(mc, md));
  mloc = fmaxf(mloc, __shfl_xor(mloc, 16, 32));    // lanes q / q+16 combine

  const float m_new = fmaxf(m_run, mloc);
  const float alpha = fast_exp2(m_run - m_new);
  #pragma unroll
  for (int gg = 0; gg < 8; ++gg) {
    e1[gg] = fast_exp2(e1[gg] - m_new);
    e2[gg] = fast_exp2(e2[gg] - m_new);
  }
  // tree sum
  const float sa = (e1[0] + e1[1]) + (e1[2] + e1[3]);
  const float sb = (e1[4] + e1[5]) + (e1[6] + e1[7]);
  const float sc = (e2[0] + e2[1]) + (e2[2] + e2[3]);
  const float sd = (e2[4] + e2[5]) + (e2[6] + e2[7]);
  float lloc = (sa + sb) + (sc + sd);
  lloc += __shfl_xor(lloc, 16, 32);
  l_run = l_run * alpha + lloc;
  m_run = m_new;
  o1 = o1 * alpha;
  o2 = o2 * alpha;

  // P^T (C layout) -> B layout.  Pack two f16 per dword, then a single
  // shfl_xor(16) pass (8 shuffles instead of 16 f32 shuffles).
  union PU { _Float16 h[2]; unsigned u; };
  unsigned h1[4], h2[4];
  #pragma unroll
  for (int j = 0; j < 4; ++j) {
    PU t1; t1.h[0] = (_Float16)e1[2 * j]; t1.h[1] = (_Float16)e1[2 * j + 1];
    PU t2; t2.h[0] = (_Float16)e2[2 * j]; t2.h[1] = (_Float16)e2[2 * j + 1];
    h1[j] = t1.u;  h2[j] = t2.u;
  }
  H16 bp;
  #pragma unroll
  for (int j = 0; j < 4; ++j) {
    const unsigned x1 = __shfl_xor(h1[j], 16, 32);
    const unsigned x2 = __shfl_xor(h2[j], 16, 32);
    bp.u[j]     = hi ? x2    : h1[j];   // keys c..c+7   / c+16..c+23
    bp.u[4 + j] = hi ? h2[j] : x1;      // keys c+8..15  / c+24..c+31
  }

  // A-matrix V^T tiles: lane m = output dim, K = tokens (contiguous in VT)
  const _Float16* v1 = vtb + (size_t)qsub * LL + c + 16 * hi;
  const _Float16* v2 = v1 + (size_t)16 * LL;
  H16 av1, av2;
  av1.part[0] = *(const h8*)v1;  av1.part[1] = *(const h8*)(v1 + 8);
  av2.part[0] = *(const h8*)v2;  av2.part[1] = *(const h8*)(v2 + 8);

  __builtin_prefetch((const void*)(v1 + 32), 0, 1);

  // O^T += V^T * P^T
  o1 = __builtin_amdgcn_wmma_f32_16x16x32_f16(false, av1.v, false, bp.v,
                                              (short)0, o1, false, false);
  o2 = __builtin_amdgcn_wmma_f32_16x16x32_f16(false, av2.v, false, bp.v,
                                              (short)0, o2, false, false);
}

// ---------------------------------------------------------------------------
// Kernel 2: flash attention with 4-way in-block split-K.
// One block = 4 waves per (b, head, 16-query tile); wave w processes a
// contiguous quarter of the key steps with private (m, l, O^T) state, then
// the partials merge through LDS (rescale by 2^(m_w - m), exp2 domain).
// ---------------------------------------------------------------------------
__global__ __launch_bounds__(32 * NSPLIT) void attn_kernel(
    const _Float16* __restrict__ Qh, const _Float16* __restrict__ Kh,
    const _Float16* __restrict__ VT, float* __restrict__ attn)
{
  __shared__ float red[NSPLIT][32][18];   // [wave][lane][o1(8) o2(8) m l]

  const int lane = threadIdx.x & 31;
  const int wv   = threadIdx.x >> 5;
  const int tile = blockIdx.x % (LL / 16);
  const int bh   = blockIdx.x / (LL / 16);
  const int b  = bh / NHEAD;
  const int hh = bh % NHEAD;
  const int qsub = lane & 15;
  const int hi   = lane >> 4;
  const int q0 = tile * 16;
  const int qi = q0 + qsub;
  const int qt = qi / NF;                     // query time block
  const int jlane = (qt + 1) * NF;            // per-lane causal horizon (excl.)

  // Q^T as B-matrix: lane q holds dims 0..7,16..23 of query q, lane q+16 dims 8..15,24..31
  const _Float16* qrow = Qh + ((size_t)bh * LL + qi) * HDIM + 8 * hi;
  H16 bq;
  bq.part[0] = *(const h8*)qrow;
  bq.part[1] = *(const h8*)(qrow + 16);

  const _Float16* kbase = Kh + (size_t)bh * LL * HDIM;
  const _Float16* vtb   = VT + (size_t)bh * (size_t)HDIM * LL;

  v8f o1 = {};            // O^T dims 0..15  (lane = query column)
  v8f o2 = {};            // O^T dims 16..31
  float m_run = -1e30f, l_run = 0.f;

  const int jraw  = ((q0 + 15) / NF + 1) * NF;   // max horizon in tile
  const int jmax  = (jraw > LL) ? LL : jraw;
  const int jfull = ((q0 / NF) + 1) * NF;        // min horizon in tile
  const int cfull = (jfull / 32) * 32;           // fully-unmasked prefix

  // contiguous quarter of the 32-key steps for this wave (bounds are *32)
  const int nsteps = (jmax + 31) / 32;
  const int cbegin = ((wv * nsteps) / NSPLIT) * 32;
  const int cend   = (wv == NSPLIT - 1) ? jmax
                                        : (((wv + 1) * nsteps) / NSPLIT) * 32;
  const int cfast  = (cend < cfull) ? cend : cfull;

  int c = cbegin;
  for (; c < cfast; c += 32)
    attn_step<false>(c, qsub, hi, jlane, kbase, vtb, bq.v, m_run, l_run, o1, o2);
  for (; c < cend; c += 32)
    attn_step<true>(c, qsub, hi, jlane, kbase, vtb, bq.v, m_run, l_run, o1, o2);

  // stash partials and merge (exp2 domain; empty waves carry m=-1e30, l=0)
  #pragma unroll
  for (int j = 0; j < 8; ++j) {
    red[wv][lane][j]     = o1[j];
    red[wv][lane][8 + j] = o2[j];
  }
  red[wv][lane][16] = m_run;
  red[wv][lane][17] = l_run;
  __syncthreads();
  if (wv != 0) return;

  float m = red[0][lane][16];
  #pragma unroll
  for (int w = 1; w < NSPLIT; ++w) m = fmaxf(m, red[w][lane][16]);
  float l = 0.f;
  float acc[16];
  #pragma unroll
  for (int j = 0; j < 16; ++j) acc[j] = 0.f;
  #pragma unroll
  for (int w = 0; w < NSPLIT; ++w) {
    const float sc = fast_exp2(red[w][lane][16] - m);
    l += red[w][lane][17] * sc;
    #pragma unroll
    for (int j = 0; j < 16; ++j) acc[j] += red[w][lane][j] * sc;
  }

  // normalize and write attn[b, qi, hh*32 + dim]
  const float inv_l = 1.f / l;
  float* orow = attn + ((size_t)b * LL + qi) * HIDN + hh * HDIM;
  float4 w0 = make_float4(acc[0] * inv_l, acc[1] * inv_l, acc[2] * inv_l, acc[3] * inv_l);
  float4 w1 = make_float4(acc[4] * inv_l, acc[5] * inv_l, acc[6] * inv_l, acc[7] * inv_l);
  float4 w2 = make_float4(acc[8] * inv_l, acc[9] * inv_l, acc[10] * inv_l, acc[11] * inv_l);
  float4 w3 = make_float4(acc[12] * inv_l, acc[13] * inv_l, acc[14] * inv_l, acc[15] * inv_l);
  *(float4*)(orow + hi * 8)          = w0;   // dims hi*8 .. hi*8+3
  *(float4*)(orow + hi * 8 + 4)      = w1;
  *(float4*)(orow + 16 + hi * 8)     = w2;   // dims 16+hi*8 ..
  *(float4*)(orow + 16 + hi * 8 + 4) = w3;
}

// ---------------------------------------------------------------------------
// Kernel 3: mean-pool over forces (commutes with linear o-proj), o-proj,
// LayerNorm(128), final 128x128 linear + SiLU.  One wave per (b, t).
// ---------------------------------------------------------------------------
__global__ __launch_bounds__(256) void final_kernel(
    const float* __restrict__ attn,
    const float* __restrict__ ow,   const float* __restrict__ ob,
    const float* __restrict__ flnw, const float* __restrict__ flnb,
    const float* __restrict__ fw,   const float* __restrict__ fb,
    float* __restrict__ out)
{
  const int lane = threadIdx.x & 31;
  const int row  = blockIdx.x * 8 + (threadIdx.x >> 5);  // [0, B*T)
  const int b = row / TT;
  const int t = row % TT;

  float pooled[4];
  #pragma unroll
  for (int i = 0; i < 4; ++i) {
    const int col = lane + 32 * i;
    float s = 0.f;
    #pragma unroll
    for (int f = 0; f < NF; ++f)
      s += attn[((size_t)b * LL + t * NF + f) * HIDN + col];
    pooled[i] = s * (1.f / NF);
  }

  // z = pooled @ o_w + o_b
  float z[4];
  #pragma unroll
  for (int i = 0; i < 4; ++i) z[i] = ob[lane + 32 * i];
  #pragma unroll
  for (int kk = 0; kk < 4; ++kk) {
    const float pv = pooled[kk];
    for (int k5 = 0; k5 < 32; ++k5) {
      const float pk = __shfl(pv, k5, 32);
      const int krow = kk * 32 + k5;
      #pragma unroll
      for (int i = 0; i < 4; ++i)
        z[i] += pk * ow[krow * HIDN + lane + 32 * i];
    }
  }

  // LayerNorm over 128
  const float mu = wsum(z[0] + z[1] + z[2] + z[3]) * (1.f / HIDN);
  float vs = 0.f;
  #pragma unroll
  for (int i = 0; i < 4; ++i) { const float d = z[i] - mu; vs += d * d; }
  const float rs = rsqrtf(wsum(vs) * (1.f / HIDN) + EPSN);
  float zn[4];
  #pragma unroll
  for (int i = 0; i < 4; ++i)
    zn[i] = (z[i] - mu) * rs * flnw[lane + 32 * i] + flnb[lane + 32 * i];

  // final = SiLU(zn @ f_w + f_b)
  float acc[4];
  #pragma unroll
  for (int i = 0; i < 4; ++i) acc[i] = fb[lane + 32 * i];
  #pragma unroll
  for (int kk = 0; kk < 4; ++kk) {
    const float zv = zn[kk];
    for (int k5 = 0; k5 < 32; ++k5) {
      const float zk = __shfl(zv, k5, 32);
      const int krow = kk * 32 + k5;
      #pragma unroll
      for (int i = 0; i < 4; ++i)
        acc[i] += zk * fw[krow * HIDN + lane + 32 * i];
    }
  }
  #pragma unroll
  for (int i = 0; i < 4; ++i) {
    const float a = acc[i];
    out[(size_t)row * FDIM + lane + 32 * i] = a / (1.f + __expf(-a));
  }
}

// ---------------------------------------------------------------------------
extern "C" void kernel_launch(void* const* d_in, const int* in_sizes, int n_in,
                              void* d_out, int out_size, void* d_ws, size_t ws_size,
                              hipStream_t stream)
{
  (void)in_sizes; (void)n_in; (void)out_size; (void)ws_size;

  const float* x     = (const float*)d_in[0];
  const float* glnw  = (const float*)d_in[1];
  const float* glnb  = (const float*)d_in[2];
  const float* gw1   = (const float*)d_in[3];
  const float* gb1   = (const float*)d_in[4];
  const float* gw2   = (const float*)d_in[5];
  const float* gb2   = (const float*)d_in[6];
  const float* qw    = (const float*)d_in[7];
  const float* qb    = (const float*)d_in[8];
  const float* kw    = (const float*)d_in[9];
  const float* kbias = (const float*)d_in[10];
  const float* vw    = (const float*)d_in[11];
  const float* vb    = (const float*)d_in[12];
  const float* ow    = (const float*)d_in[13];
  const float* ob    = (const float*)d_in[14];
  const float* flnw  = (const float*)d_in[15];
  const float* flnb  = (const float*)d_in[16];
  const float* fw    = (const float*)d_in[17];
  const float* fb    = (const float*)d_in[18];
  // d_in[19] == T (compile-time constant here)

  // Workspace layout: Qh | Kh | VT (f16) | attn (f32)   (~7.9 MB total)
  const size_t qkvElems = (size_t)BB * NHEAD * LL * HDIM;   // 786432
  _Float16* Qh = (_Float16*)d_ws;
  _Float16* Kh = Qh + qkvElems;
  _Float16* VT = Kh + qkvElems;
  float*    at = (float*)(VT + qkvElems);

  prep_kernel<<<BB * LL / 8, 256, 0, stream>>>(
      x, glnw, glnb, gw1, gb1, gw2, gb2, qw, qb, kw, kbias, vw, vb, Qh, Kh, VT);

  attn_kernel<<<BB * NHEAD * (LL / 16), 32 * NSPLIT, 0, stream>>>(Qh, Kh, VT, at);

  final_kernel<<<BB * TT / 8, 256, 0, stream>>>(
      at, ow, ob, flnw, flnb, fw, fb, (float*)d_out);
}
